// LTAE_88562225643738
// MI455X (gfx1250) — matrix operand order repaired and checked by reading, not compiled, not run on hardware
//
#include <hip/hip_runtime.h>
#include <hip/hip_bf16.h>

typedef __bf16 bf16_t;
typedef __attribute__((ext_vector_type(16))) __bf16 v16bf;
typedef __attribute__((ext_vector_type(8)))  float  v8f;
typedef __attribute__((ext_vector_type(4)))  unsigned int su4;   // TDM D# group0 (4 SGPRs)
typedef __attribute__((ext_vector_type(8)))  unsigned int su8;   // TDM D# group1 (8 SGPRs)

#define EPS 1e-5f
#define INVSQ8 0.35355339059327f   // 1/sqrt(8)
#define PE_K  0.02698342692f       // ln(1000)/256

// ---------------- WMMA helpers (CDNA5 16x16x32 bf16) ----------------

static __device__ inline v8f wmma_bf16(v8f acc, v16bf a, v16bf b) {
    // (neg_a, A, neg_b, B, c_mod, C, reuse_a, reuse_b)
    return __builtin_amdgcn_wmma_f32_16x16x32_bf16(false, a, false, b, (short)0, acc, false, false);
}

// A fragment: 16x32 bf16 tile, source row-major [M][stride] in LDS.
static __device__ inline v16bf load_a_lds(const bf16_t* base, int stride, int mbase, int kt) {
    int lane = threadIdx.x & 31;
    int r  = mbase + (lane & 15);
    int k0 = kt * 32 + ((lane >> 4) << 4);
    const bf16_t* p = base + r * stride + k0;
    union { v16bf v; uint4 q[2]; } u;
    u.q[0] = *(const uint4*)(p);
    u.q[1] = *(const uint4*)(p + 8);
    return u.v;
}

// B fragment: 32x16 bf16, source stored as [N][K] row-major, LDS bf16.
static __device__ inline v16bf load_b_lds(const bf16_t* Wt, int ldk, int nbase, int kt) {
    int lane = threadIdx.x & 31;
    int n  = nbase + (lane & 15);
    int k0 = kt * 32 + ((lane >> 4) << 4);
    const bf16_t* p = Wt + n * ldk + k0;
    union { v16bf v; uint4 q[2]; } u;
    u.q[0] = *(const uint4*)(p);
    u.q[1] = *(const uint4*)(p + 8);
    return u.v;
}

static __device__ inline v16bf pack16(float4 q0, float4 q1, float4 q2, float4 q3) {
    v16bf f;
    f[0]=(bf16_t)q0.x;  f[1]=(bf16_t)q0.y;  f[2]=(bf16_t)q0.z;  f[3]=(bf16_t)q0.w;
    f[4]=(bf16_t)q1.x;  f[5]=(bf16_t)q1.y;  f[6]=(bf16_t)q1.z;  f[7]=(bf16_t)q1.w;
    f[8]=(bf16_t)q2.x;  f[9]=(bf16_t)q2.y;  f[10]=(bf16_t)q2.z; f[11]=(bf16_t)q2.w;
    f[12]=(bf16_t)q3.x; f[13]=(bf16_t)q3.y; f[14]=(bf16_t)q3.z; f[15]=(bf16_t)q3.w;
    return f;
}

// B fragment converted on the fly from f32 global weights [N][K] (64B-aligned rows).
static __device__ inline v16bf load_b_g(const float* __restrict__ W, int ldk, int nbase, int kt) {
    int lane = threadIdx.x & 31;
    int n  = nbase + (lane & 15);
    int k0 = kt * 32 + ((lane >> 4) << 4);
    const float4* p = (const float4*)(W + n * ldk + k0);
    return pack16(p[0], p[1], p[2], p[3]);
}

// Guarded (zero-padded rows) variant for the 20x32 Wd3.
static __device__ inline v16bf load_b_g_guard(const float* __restrict__ W, int ldk, int nbase,
                                              int kt, int nmax) {
    int lane = threadIdx.x & 31;
    int n  = nbase + (lane & 15);
    int k0 = kt * 32 + ((lane >> 4) << 4);
    if (n < nmax) {
        const float4* p = (const float4*)(W + n * ldk + k0);
        return pack16(p[0], p[1], p[2], p[3]);
    }
    v16bf f;
#pragma unroll
    for (int i = 0; i < 16; ++i) f[i] = (bf16_t)0.0f;
    return f;
}

// ---- Tensor Data Mover: 1D/padded tile load, global -> LDS (CDNA5 ISA) ----
// g1w0 extra bits allow pad_enable/pad_interval/pad_amount. nelem = dwords.
static __device__ inline void tdm_load_1d(unsigned lds_off, unsigned long long gaddr,
                                          unsigned nelem_dw, unsigned g1w0_extra) {
    su4 g0;
    g0[0] = 1u;                                    // count=1 valid descriptor, user mode
    g0[1] = lds_off;                               // LDS byte address (dynamic base == 0)
    g0[2] = (unsigned)(gaddr & 0xFFFFFFFFu);       // global_addr[31:0]
    g0[3] = (unsigned)((gaddr >> 32) & 0x1FFFFFFu) // global_addr[56:32]
            | (2u << 30);                          // type = 2 ("image")
    su8 g1;
    g1[0] = (2u << 16) | g1w0_extra;               // data_size = 4 bytes (+ pad ctl)
    g1[1] = (nelem_dw & 0xFFFFu) << 16;            // tensor_dim0[15:0]   (atomic addr = 0)
    g1[2] = (nelem_dw >> 16) | (1u << 16);         // tensor_dim0[31:16], tensor_dim1 = 1
    g1[3] = nelem_dw << 16;                        // tile_dim0 = nelem   (tensor_dim1 hi = 0)
    g1[4] = 0u;                                    // tile_dim1/2 unused
    g1[5] = nelem_dw;                              // tensor_dim0_stride
    g1[6] = 0u;
    g1[7] = 0u;
    asm volatile("tensor_load_to_lds %0, %1" :: "s"(g0), "s"(g1) : "memory");
}

// ============================ Kernel 1 ==============================
// Per block: 8 samples (192 e-rows). TDM-stage x slab -> LN -> conv(+BN1 folded)
// -> +PE -> e (bf16 LDS) -> logits = e x qkw^T via WMMA (Q folded into Wk)
// -> softmax(S=24) -> o = attn*v -> o[B,256] bf16 to workspace.

#define K1_ES   264                       // padded e stride (bf16): 528B = 16B mod 256
#define K1_OFF_E    0
#define K1_OFF_QKW  101376                // 192*264*2
#define K1_OFF_LOG  109824                // + 16*264*2
#define K1_OFF_CWF  122112                // + 8*16*24*4
#define K1_OFF_T1   132352                // + 256*10*4
#define K1_OFF_QB   133376                // + 256*4
#define K1_OFF_LNG  133440
#define K1_OFF_LNB  133504
#define K1_OFF_DIV  133568
#define K1_OFF_X    134080                // + 128*4 ; x slab 192*11*4 = 8448B
#define K1_SMEM     142528

__global__ __launch_bounds__(256)
void ltae_stage1(const float* __restrict__ x,
                 const float* __restrict__ ln1_g, const float* __restrict__ ln1_b,
                 const float* __restrict__ conv_w, const float* __restrict__ conv_b,
                 const float* __restrict__ bn1_g, const float* __restrict__ bn1_b,
                 const float* __restrict__ bn1_m, const float* __restrict__ bn1_v,
                 const float* __restrict__ Q, const float* __restrict__ Wk,
                 const float* __restrict__ bk,
                 bf16_t* __restrict__ o_out) {
    extern __shared__ __align__(16) char smem[];
    bf16_t* e_lds  = (bf16_t*)(smem + K1_OFF_E);     // [192][264]
    bf16_t* qkw    = (bf16_t*)(smem + K1_OFF_QKW);   // [16][264] (Q folded, *1/sqrt8)
    float*  logits = (float* )(smem + K1_OFF_LOG);   // [8][16][24]
    float*  cwf    = (float* )(smem + K1_OFF_CWF);   // [256][10] conv_w * bn1 scale
    float*  t1     = (float* )(smem + K1_OFF_T1);    // [256] folded bias
    float*  qb     = (float* )(smem + K1_OFF_QB);    // [16]
    float*  lng    = (float* )(smem + K1_OFF_LNG);   // [10]
    float*  lnb    = (float* )(smem + K1_OFF_LNB);   // [10]
    float*  divt   = (float* )(smem + K1_OFF_DIV);   // [128] PE frequencies
    float*  x_lds  = (float* )(smem + K1_OFF_X);     // [192][11]

    const int t    = threadIdx.x;
    const int wave = t >> 5;
    const int lane = t & 31;

    // ---- async: DMA this block's x slab (8 samples x 24 x 11 f32) into LDS ----
    if (wave == 0) {
        unsigned long long ga =
            (unsigned long long)(size_t)(x + (size_t)blockIdx.x * (8 * 24 * 11));
        tdm_load_1d(K1_OFF_X, ga, 8 * 24 * 11, 0u);
    }

    // ---- init (overlaps TDM): fold Q into Wk, BN1 into conv, PE table ----
    {
        int h  = t >> 4;
        int d0 = (t & 15) << 4;
        float qrow[8];
#pragma unroll
        for (int k = 0; k < 8; ++k) qrow[k] = Q[h * 8 + k];
#pragma unroll
        for (int j = 0; j < 16; ++j) {
            int d = d0 + j;
            float a = 0.f;
#pragma unroll
            for (int k = 0; k < 8; ++k) a += qrow[k] * Wk[(h * 8 + k) * 256 + d];
            qkw[h * K1_ES + d] = (bf16_t)(a * INVSQ8);
        }
        if ((t & 15) == 0) {
            float a = 0.f;
#pragma unroll
            for (int k = 0; k < 8; ++k) a += qrow[k] * bk[h * 8 + k];
            qb[h] = a * INVSQ8;
        }
        int d = t;
        float s = bn1_g[d] * rsqrtf(bn1_v[d] + EPS);
#pragma unroll
        for (int c = 0; c < 10; ++c) cwf[d * 10 + c] = conv_w[d * 10 + c] * s;
        t1[d] = (conv_b[d] - bn1_m[d]) * s + bn1_b[d];
        if (t < 128) divt[t] = __expf(-(float)(2 * t) * PE_K);
        if (t < 10)  { lng[t] = ln1_g[t]; lnb[t] = ln1_b[t]; }
    }
    if (wave == 0) __builtin_amdgcn_s_wait_tensorcnt(0);
    __syncthreads();

    // ---- phase 1: e rows (one row per wave iteration, 8 channels/lane) ----
    for (int r = wave; r < 192; r += 8) {
        const float* xp = x_lds + r * 11;
        float pos = xp[0];
        float f[10];
        float mu = 0.f;
#pragma unroll
        for (int c = 0; c < 10; ++c) { f[c] = xp[1 + c]; mu += f[c]; }
        mu *= 0.1f;
        float var = 0.f;
#pragma unroll
        for (int c = 0; c < 10; ++c) { float d = f[c] - mu; var += d * d; }
        var *= 0.1f;
        float inv = rsqrtf(var + EPS);
        float nf[10];
#pragma unroll
        for (int c = 0; c < 10; ++c) nf[c] = (f[c] - mu) * inv * lng[c] + lnb[c];

        int d0 = lane << 3;
        union { bf16_t h[8]; uint4 q; } ov;
#pragma unroll
        for (int j = 0; j < 4; ++j) {
            int de = d0 + 2 * j;
            float ae = t1[de], ao = t1[de + 1];
#pragma unroll
            for (int c = 0; c < 10; ++c) {
                ae += cwf[de * 10 + c] * nf[c];
                ao += cwf[(de + 1) * 10 + c] * nf[c];
            }
            float sv, cv;
            __sincosf(pos * divt[de >> 1], &sv, &cv);
            ov.h[2 * j]     = (bf16_t)(ae + sv);
            ov.h[2 * j + 1] = (bf16_t)(ao + cv);
        }
        *(uint4*)(e_lds + r * K1_ES + d0) = ov.q;
    }
    __syncthreads();

    // ---- phase 2: logits[192][16] = e(192x256) x qkw^T(256x16), WMMA ----
    for (int tile = wave; tile < 12; tile += 8) {
        v8f acc = {};
#pragma unroll
        for (int kt = 0; kt < 8; ++kt) {
            v16bf a = load_a_lds(e_lds, K1_ES, tile * 16, kt);
            v16bf b = load_b_lds(qkw,  K1_ES, 0, kt);
            acc = wmma_bf16(acc, a, b);
        }
        int n = lane & 15;          // head index
        int moff = (lane >> 4) << 3;
        float qbv = qb[n];
#pragma unroll
        for (int i = 0; i < 8; ++i) {
            int row = tile * 16 + moff + i;
            int bl = row / 24, s = row - bl * 24;
            logits[(bl * 16 + n) * 24 + s] = acc[i] + qbv;
        }
    }
    __syncthreads();

    // ---- phase 3: softmax over S=24, one (b,h) per thread ----
    if (t < 128) {
        float* lp = logits + t * 24;
        float mx = lp[0];
#pragma unroll
        for (int s = 1; s < 24; ++s) mx = fmaxf(mx, lp[s]);
        float sum = 0.f;
        float ex[24];
#pragma unroll
        for (int s = 0; s < 24; ++s) { ex[s] = __expf(lp[s] - mx); sum += ex[s]; }
        float rs = 1.f / sum;
#pragma unroll
        for (int s = 0; s < 24; ++s) lp[s] = ex[s] * rs;
    }
    __syncthreads();

    // ---- phase 4: o[b][d] = sum_s attn[b][d>>4][s] * e[b*24+s][d] ----
    {
        int b = t >> 5;
        int d0 = lane << 3;
        long gb = (long)blockIdx.x * 8 + b;
        union { bf16_t h[8]; uint4 q; } ov;
#pragma unroll
        for (int j = 0; j < 8; ++j) {
            int d = d0 + j;
            const float* ap = logits + (b * 16 + (d >> 4)) * 24;
            const bf16_t* ep = e_lds + (b * 24) * K1_ES + d;
            float acc = 0.f;
#pragma unroll
            for (int s = 0; s < 24; ++s) acc += ap[s] * (float)ep[s * K1_ES];
            ov.h[j] = (bf16_t)acc;
        }
        *(uint4*)(o_out + gb * 256 + d0) = ov.q;
    }
}

// ============================ Kernel 2 ==============================
// MLP head for 64 rows/block: o(256) ->W1+BN2+ReLU-> 128 ->LN-> Wd1+BN+ReLU -> 64
// -> Wd2+BN+ReLU -> 32 -> Wd3(pad to 32) -> out 20.  All GEMMs via WMMA bf16.
// o tile staged by TDM with hardware LDS padding (stride 264 bf16).

#define K2_OFF_A    0
#define K2_OFF_M32  33792                 // 64*264*2
#define K2_OFF_MBF  67584                 // + 64*132*4
#define K2_OFF_D1   84992                 // + 64*136*2
#define K2_OFF_D2   94208                 // + 64*72*2
#define K2_OFF_S2   99328                 // + 64*40*2
#define K2_OFF_T2   99840
#define K2_OFF_SD1  100352
#define K2_OFF_TD1  100608
#define K2_OFF_SD2  100864
#define K2_OFF_TD2  100992
#define K2_SMEM     101120

__global__ __launch_bounds__(256)
void ltae_stage2(const bf16_t* __restrict__ o_in,
                 const float* __restrict__ W1,  const float* __restrict__ b1,
                 const float* __restrict__ bn2_g, const float* __restrict__ bn2_b,
                 const float* __restrict__ bn2_m, const float* __restrict__ bn2_v,
                 const float* __restrict__ oln_g, const float* __restrict__ oln_b,
                 const float* __restrict__ Wd1, const float* __restrict__ bd1,
                 const float* __restrict__ bnd1_g, const float* __restrict__ bnd1_b,
                 const float* __restrict__ bnd1_m, const float* __restrict__ bnd1_v,
                 const float* __restrict__ Wd2, const float* __restrict__ bd2,
                 const float* __restrict__ bnd2_g, const float* __restrict__ bnd2_b,
                 const float* __restrict__ bnd2_m, const float* __restrict__ bnd2_v,
                 const float* __restrict__ Wd3, const float* __restrict__ bd3,
                 float* __restrict__ out) {
    extern __shared__ __align__(16) char smem[];
    bf16_t* a_lds = (bf16_t*)(smem + K2_OFF_A);     // [64][264] (TDM pads rows)
    float*  m32   = (float* )(smem + K2_OFF_M32);   // [64][132] pre-LN activations
    bf16_t* m_bf  = (bf16_t*)(smem + K2_OFF_MBF);   // [64][136]
    bf16_t* d1l   = (bf16_t*)(smem + K2_OFF_D1);    // [64][72]
    bf16_t* d2l   = (bf16_t*)(smem + K2_OFF_D2);    // [64][40]
    float*  s2    = (float* )(smem + K2_OFF_S2);
    float*  t2    = (float* )(smem + K2_OFF_T2);
    float*  sd1   = (float* )(smem + K2_OFF_SD1);
    float*  td1   = (float* )(smem + K2_OFF_TD1);
    float*  sd2   = (float* )(smem + K2_OFF_SD2);
    float*  td2   = (float* )(smem + K2_OFF_TD2);

    const int t    = threadIdx.x;
    const int wave = t >> 5;
    const int lane = t & 31;
    const long row0 = (long)blockIdx.x * 64;

    // ---- async: DMA o tile (64x256 bf16 = 32KB) with 16B pad per 512B row ----
    if (wave == 0) {
        unsigned long long ga = (unsigned long long)(size_t)(o_in + row0 * 256);
        // pad_enable | pad_interval=6 (128 DW = one 512B row) | pad_amount=3 (4 DW = 16B)
        tdm_load_1d(K2_OFF_A, ga, (64 * 256 * 2) / 4,
                    (1u << 20) | (6u << 22) | (3u << 25));
    }

    // L2 warm-up for the shared weight matrices (global_prefetch_b8).
    __builtin_prefetch((const char*)W1  + t * 512, 0, 1);   // 128 KB
    __builtin_prefetch((const char*)Wd1 + t * 128, 0, 1);   // 32 KB

    // ---- fold BN params (overlaps TDM) ----
    if (t < 128) {
        float s = bn2_g[t] * rsqrtf(bn2_v[t] + EPS);
        s2[t] = s;  t2[t] = (b1[t] - bn2_m[t]) * s + bn2_b[t];
    }
    if (t < 64) {
        float s = bnd1_g[t] * rsqrtf(bnd1_v[t] + EPS);
        sd1[t] = s; td1[t] = (bd1[t] - bnd1_m[t]) * s + bnd1_b[t];
    }
    if (t < 32) {
        float s = bnd2_g[t] * rsqrtf(bnd2_v[t] + EPS);
        sd2[t] = s; td2[t] = (bd2[t] - bnd2_m[t]) * s + bnd2_b[t];
    }
    if (wave == 0) __builtin_amdgcn_s_wait_tensorcnt(0);
    __syncthreads();

    // ---- GEMM1: (64x256) x W1^T(256x128) + BN2 + ReLU -> m32 ----
    for (int id = wave; id < 32; id += 8) {
        int mt = id >> 3, nt = id & 7;
        v8f acc = {};
#pragma unroll
        for (int kt = 0; kt < 8; ++kt) {
            v16bf a = load_a_lds(a_lds, 264, mt * 16, kt);
            v16bf b = load_b_g(W1, 256, nt * 16, kt);
            acc = wmma_bf16(acc, a, b);
        }
        int n = nt * 16 + (lane & 15);
        int moff = (lane >> 4) << 3;
        float sv = s2[n], tv = t2[n];
#pragma unroll
        for (int i = 0; i < 8; ++i) {
            int row = mt * 16 + moff + i;
            m32[row * 132 + n] = fmaxf(acc[i] * sv + tv, 0.f);
        }
    }
    __syncthreads();

    // ---- LayerNorm(128) per row -> m_bf ----
    if (t < 64) {
        const float* rp = m32 + t * 132;
        float mu = 0.f;
        for (int c = 0; c < 128; ++c) mu += rp[c];
        mu *= (1.f / 128.f);
        float var = 0.f;
        for (int c = 0; c < 128; ++c) { float d = rp[c] - mu; var += d * d; }
        var *= (1.f / 128.f);
        float inv = rsqrtf(var + EPS);
        bf16_t* dp = m_bf + t * 136;
        for (int c = 0; c < 128; ++c)
            dp[c] = (bf16_t)((rp[c] - mu) * inv * oln_g[c] + oln_b[c]);
    }
    __syncthreads();

    // ---- GEMM2: (64x128) x Wd1^T(128x64) + BN + ReLU -> d1 ----
    for (int id = wave; id < 16; id += 8) {
        int mt = id >> 2, nt = id & 3;
        v8f acc = {};
#pragma unroll
        for (int kt = 0; kt < 4; ++kt) {
            v16bf a = load_a_lds(m_bf, 136, mt * 16, kt);
            v16bf b = load_b_g(Wd1, 128, nt * 16, kt);
            acc = wmma_bf16(acc, a, b);
        }
        int n = nt * 16 + (lane & 15);
        int moff = (lane >> 4) << 3;
        float sv = sd1[n], tv = td1[n];
#pragma unroll
        for (int i = 0; i < 8; ++i) {
            int row = mt * 16 + moff + i;
            d1l[row * 72 + n] = (bf16_t)fmaxf(acc[i] * sv + tv, 0.f);
        }
    }
    __syncthreads();

    // ---- GEMM3: (64x64) x Wd2^T(64x32) + BN + ReLU -> d2 ----
    {
        int mt = wave >> 1, nt = wave & 1;
        v8f acc = {};
#pragma unroll
        for (int kt = 0; kt < 2; ++kt) {
            v16bf a = load_a_lds(d1l, 72, mt * 16, kt);
            v16bf b = load_b_g(Wd2, 64, nt * 16, kt);
            acc = wmma_bf16(acc, a, b);
        }
        int n = nt * 16 + (lane & 15);
        int moff = (lane >> 4) << 3;
        float sv = sd2[n], tv = td2[n];
#pragma unroll
        for (int i = 0; i < 8; ++i) {
            int row = mt * 16 + moff + i;
            d2l[row * 40 + n] = (bf16_t)fmaxf(acc[i] * sv + tv, 0.f);
        }
    }
    __syncthreads();

    // ---- GEMM4: (64x32) x Wd3^T(32x20, zero-padded to 32) -> out ----
    {
        int mt = wave >> 1, nt = wave & 1;
        v8f acc = {};
        v16bf a = load_a_lds(d2l, 40, mt * 16, 0);
        v16bf b = load_b_g_guard(Wd3, 32, nt * 16, 0, 20);
        acc = wmma_bf16(acc, a, b);
        int n = nt * 16 + (lane & 15);
        int moff = (lane >> 4) << 3;
        if (n < 20) {
            float bv = bd3[n];
#pragma unroll
            for (int i = 0; i < 8; ++i) {
                int row = mt * 16 + moff + i;
                out[(row0 + row) * 20 + n] = acc[i] + bv;
            }
        }
    }
}

// ============================ launcher ==============================

extern "C" void kernel_launch(void* const* d_in, const int* in_sizes, int n_in,
                              void* d_out, int out_size, void* d_ws, size_t ws_size,
                              hipStream_t stream) {
    (void)in_sizes; (void)n_in; (void)out_size; (void)ws_size;
    const float* x      = (const float*)d_in[0];
    const float* ln1_g  = (const float*)d_in[1];
    const float* ln1_b  = (const float*)d_in[2];
    const float* conv_w = (const float*)d_in[3];
    const float* conv_b = (const float*)d_in[4];
    const float* bn1_g  = (const float*)d_in[5];
    const float* bn1_b  = (const float*)d_in[6];
    const float* bn1_m  = (const float*)d_in[7];
    const float* bn1_v  = (const float*)d_in[8];
    const float* Q      = (const float*)d_in[9];
    const float* Wk     = (const float*)d_in[10];
    const float* bk     = (const float*)d_in[11];
    const float* W1     = (const float*)d_in[12];
    const float* b1     = (const float*)d_in[13];
    const float* bn2_g  = (const float*)d_in[14];
    const float* bn2_b  = (const float*)d_in[15];
    const float* bn2_m  = (const float*)d_in[16];
    const float* bn2_v  = (const float*)d_in[17];
    const float* oln_g  = (const float*)d_in[18];
    const float* oln_b  = (const float*)d_in[19];
    const float* Wd1    = (const float*)d_in[20];
    const float* bd1    = (const float*)d_in[21];
    const float* bnd1_g = (const float*)d_in[22];
    const float* bnd1_b = (const float*)d_in[23];
    const float* bnd1_m = (const float*)d_in[24];
    const float* bnd1_v = (const float*)d_in[25];
    const float* Wd2    = (const float*)d_in[26];
    const float* bd2    = (const float*)d_in[27];
    const float* bnd2_g = (const float*)d_in[28];
    const float* bnd2_b = (const float*)d_in[29];
    const float* bnd2_m = (const float*)d_in[30];
    const float* bnd2_v = (const float*)d_in[31];
    const float* Wd3    = (const float*)d_in[32];
    const float* bd3    = (const float*)d_in[33];

    bf16_t* o_ws = (bf16_t*)d_ws;              // [16384][256] bf16 = 8 MB
    float*  outp = (float*)d_out;              // [16384][20]

    ltae_stage1<<<16384 / 8, 256, K1_SMEM, stream>>>(
        x, ln1_g, ln1_b, conv_w, conv_b, bn1_g, bn1_b, bn1_m, bn1_v,
        Q, Wk, bk, o_ws);

    ltae_stage2<<<16384 / 64, 256, K2_SMEM, stream>>>(
        o_ws, W1, b1, bn2_g, bn2_b, bn2_m, bn2_v, oln_g, oln_b,
        Wd1, bd1, bnd1_g, bnd1_b, bnd1_m, bnd1_v,
        Wd2, bd2, bnd2_g, bnd2_b, bnd2_m, bnd2_v,
        Wd3, bd3, outp);
}
